// ModConv1x1_59081570125044
// MI455X (gfx1250) — matrix operand order repaired and checked by reading, not compiled, not run on hardware
//
#include <hip/hip_runtime.h>

typedef __attribute__((ext_vector_type(16))) _Float16 v16h;
typedef __attribute__((ext_vector_type(8)))  _Float16 v8h;
typedef __attribute__((ext_vector_type(8)))  float    v8f;

#define B_SZ   8
#define C_CH   128
#define H_SZ   256
#define W_SZ   256
#define HW_SZ  (H_SZ * W_SZ)
#define OUT_C  64
#define IN_C   (C_CH * 9)   // 1152
#define KC     32           // channels per K-chunk
#define NQ     (C_CH / KC)  // 4 chunks per tap

// One block (8 wave32) computes out[b, 0:64, y, 0:256].
// Tap-decomposed GEMM: out[o,p] = sum_{tap} sum_c wmod[o,c,tap] * x[b,c,y+kh-1,p+kw-1].
// Per tap the B-fragment element offset is e*H*W -> folds into the load immediate.
// Split-f16 (Markidis) WMMA keeps fp32-level accuracy: Ah*Bh + Ah*Bl + Al*Bh.
// Boundary handling: clamp the per-lane column so every load is in-bounds and
// branch-free, then zero OOB lanes with one v_cndmask per element.
__global__ __launch_bounds__(256) void modconv3x3_wmma_kernel(
    const float* __restrict__ x,      // (B, C, H, W)
    const float* __restrict__ style,  // (B, IN_C)
    const float* __restrict__ weight, // (OUT_C, IN_C)
    float* __restrict__ out)          // (B, OUT_C, H, W)
{
    // Fragment-ready A staging: [q][mtile][lane][e] halves (hi and lo parts)
    __shared__ _Float16 aH[NQ * 4 * 32 * 16];   // 16 KB
    __shared__ _Float16 aL[NQ * 4 * 32 * 16];   // 16 KB

    const int y      = blockIdx.x;
    const int b      = blockIdx.y;
    const int t      = threadIdx.x;
    const int lane   = t & 31;
    const int wv     = t >> 5;        // wave 0..7
    const int laneLo = lane & 15;     // N column within tile
    const int laneHi = lane >> 4;     // 0/1: K/M sub-block select per ISA layout
    const int x0     = wv * 32;       // this wave's pixel base

    // Staging: thread t handles out-channel so, channel range sk0..sk0+7 (per chunk).
    // Fragment layout inverse: lane=(o&15)+(k>>4)*16, e=k&15.
    // k>>3 and the e-base are constant across j -> 8 consecutive halves (16B store).
    const int so    = t >> 2;                                   // 0..63
    const int sk0   = (t & 3) * 8;                              // 0,8,16,24
    const int sm    = so >> 4;
    const int slane = (so & 15) + ((sk0 >> 4) << 4);
    const int sidx  = ((sm * 32 + slane) << 4) + (sk0 & 15);

    v8f acc[4][2];
    #pragma unroll
    for (int m = 0; m < 4; ++m)
        #pragma unroll
        for (int n = 0; n < 2; ++n)
            #pragma unroll
            for (int r = 0; r < 8; ++r)
                acc[m][n][r] = 0.0f;

    for (int tap = 0; tap < 9; ++tap) {
        const int kh = tap / 3;
        const int kw = tap - kh * 3;
        const int yy = y + kh - 1;
        if (yy < 0 || yy >= H_SZ) continue;   // block-uniform: skip border taps

        // Per-lane xx validity: only px 0 (kw=0) / px 255 (kw=2) can be OOB.
        // Clamp the column so addresses are always valid -> unconditional loads.
        const int  xxr0 = x0 + laneLo + kw - 1;
        const int  xxr1 = xxr0 + 16;
        const bool ok0  = (xxr0 >= 0) && (xxr0 < W_SZ);
        const bool ok1  = (xxr1 >= 0) && (xxr1 < W_SZ);
        const int  xxc0 = ok0 ? xxr0 : 0;
        const int  xxc1 = ok1 ? xxr1 : 0;

        __syncthreads();  // previous tap's fragment reads done

        // ---- Stage this tap's modulated weight (64 oc x 128 ch) into LDS ----
        #pragma unroll
        for (int q = 0; q < NQ; ++q) {
            v8h hv, lv;
            #pragma unroll
            for (int j = 0; j < 8; ++j) {
                const int kg = (q * KC + sk0 + j) * 9 + tap;
                const float wm = weight[(size_t)so * IN_C + kg]
                               * style[(size_t)b * IN_C + kg];
                const _Float16 hi = (_Float16)wm;
                hv[j] = hi;
                lv[j] = (_Float16)(wm - (float)hi);
            }
            *(v8h*)&aH[q * 2048 + sidx] = hv;
            *(v8h*)&aL[q * 2048 + sidx] = lv;
        }

        __syncthreads();  // staging visible to all waves

        for (int q = 0; q < NQ; ++q) {
            // ---- Load A fragments (32B contiguous per lane) ----
            v16h Ah[4], Al[4];
            #pragma unroll
            for (int m = 0; m < 4; ++m) {
                Ah[m] = *(const v16h*)&aH[q * 2048 + ((m * 32 + lane) << 4)];
                Al[m] = *(const v16h*)&aL[q * 2048 + ((m * 32 + lane) << 4)];
            }

            // Per-lane x base: channel q*KC + 16*laneHi, row yy.
            const float* xb = x + (((size_t)b * C_CH + q * KC + laneHi * 16) * H_SZ + yy) * W_SZ;

            #pragma unroll
            for (int n = 0; n < 2; ++n) {
                const bool ok = (n == 0) ? ok0  : ok1;
                const float* xp = xb + ((n == 0) ? xxc0 : xxc1);

                // B layout: lane holds column N=lane&15, K = 16*laneHi + e.
                // Element e -> channel +e -> byte offset e*HW*4 (load immediate).
                // Unconditional loads (clamped address), then zero OOB lanes.
                float vraw[16];
                #pragma unroll
                for (int e = 0; e < 16; ++e)
                    vraw[e] = xp[(size_t)e * HW_SZ];

                v16h Bh, Bl;
                #pragma unroll
                for (int e = 0; e < 16; ++e) {
                    const float v = ok ? vraw[e] : 0.0f;
                    const _Float16 h = (_Float16)v;
                    Bh[e] = h;
                    Bl[e] = (_Float16)(v - (float)h);
                }

                #pragma unroll
                for (int m = 0; m < 4; ++m) {
                    acc[m][n] = __builtin_amdgcn_wmma_f32_16x16x32_f16(
                        false, Ah[m], false, Bh, (short)0, acc[m][n], false, false);
                    acc[m][n] = __builtin_amdgcn_wmma_f32_16x16x32_f16(
                        false, Ah[m], false, Bl, (short)0, acc[m][n], false, false);
                    acc[m][n] = __builtin_amdgcn_wmma_f32_16x16x32_f16(
                        false, Al[m], false, Bh, (short)0, acc[m][n], false, false);
                }
            }
        }
    }

    // ---- Write D: VGPR r -> M = r + 8*laneHi, N = laneLo ----
    // Non-temporal: out is write-once, keep L2 for the shared x rows.
    #pragma unroll
    for (int m = 0; m < 4; ++m) {
        #pragma unroll
        for (int n = 0; n < 2; ++n) {
            const int px = x0 + n * 16 + laneLo;
            #pragma unroll
            for (int r = 0; r < 8; ++r) {
                const int oc = m * 16 + laneHi * 8 + r;
                __builtin_nontemporal_store(
                    acc[m][n][r],
                    &out[(((size_t)b * OUT_C + oc) * H_SZ + y) * W_SZ + px]);
            }
        }
    }
}

extern "C" void kernel_launch(void* const* d_in, const int* in_sizes, int n_in,
                              void* d_out, int out_size, void* d_ws, size_t ws_size,
                              hipStream_t stream) {
    (void)in_sizes; (void)n_in; (void)d_ws; (void)ws_size; (void)out_size;
    const float* x      = (const float*)d_in[0];
    const float* style  = (const float*)d_in[1];
    const float* weight = (const float*)d_in[2];
    float* out          = (float*)d_out;

    dim3 grid(H_SZ, B_SZ);   // one block per (row y, batch b)
    dim3 block(256);         // 8 wave32
    modconv3x3_wmma_kernel<<<grid, block, 0, stream>>>(x, style, weight, out);
}
